// Head_47906065219748
// MI455X (gfx1250) — compile-verified, matrix operand order checked by reading, not run on hardware
//
#include <hip/hip_runtime.h>
#include <stdint.h>

// Problem constants (match reference).
#define BB  4
#define TT  4096
#define DD  1024
#define HSZ 64

// Padded LDS row strides (halves): conflict-free on 64 banks, 16B-aligned.
#define WSTR 56   // weight tiles: 112B row stride, lane stride 28 banks
#define KSTR 72   // K/V/P tiles: 144B row stride, lane stride 36 banks

typedef __attribute__((ext_vector_type(16))) _Float16 v16h;
typedef __attribute__((ext_vector_type(8)))  _Float16 v8h;
typedef __attribute__((ext_vector_type(8)))  float    v8f;
typedef __attribute__((ext_vector_type(4)))  float    v4f;

static __device__ __forceinline__ v16h make_frag(v8h lo, v8h hi) {
    v16h r;
#pragma unroll
    for (int i = 0; i < 8; i++) { r[i] = lo[i]; r[8 + i] = hi[i]; }
    return r;
}

// CDNA5 async copy: global -> LDS, 16 bytes per lane, tracked by ASYNCcnt.
static __device__ __forceinline__ void async_load_b128(uint32_t lds_off, uint64_t gaddr) {
    asm volatile("global_load_async_to_lds_b128 %0, %1, off"
                 :: "v"(lds_off), "v"(gaddr) : "memory");
}
static __device__ __forceinline__ void wait_async0() {
    asm volatile("s_wait_asynccnt 0x0" ::: "memory");
}

// ---------------------------------------------------------------------------
// Kernel 0: convert Wk / Wv (f32, [HS, D] row-major) to f16 once.
// ---------------------------------------------------------------------------
__global__ void cvt_w_kernel(const float* __restrict__ Wk,
                             const float* __restrict__ Wv,
                             _Float16* __restrict__ Wkh,
                             _Float16* __restrict__ Wvh) {
    int i = blockIdx.x * blockDim.x + threadIdx.x;
    if (i < HSZ * DD) {
        Wkh[i] = (_Float16)Wk[i];
        Wvh[i] = (_Float16)Wv[i];
    }
}

// ---------------------------------------------------------------------------
// Kernel 1: projections K = x*Wk^T, V(=q) = x*Wv^T via WMMA f16 -> f32.
// 256 threads = 8 waves/block; each wave computes a 16x64 tile of both K and V.
// Weight k-chunks (shared by all 8 waves) are double-buffered into LDS with
// GLOBAL_LOAD_ASYNC_TO_LDS_B128; x is software-pipelined through registers.
// ---------------------------------------------------------------------------
__global__ void __launch_bounds__(256)
proj_kernel(const float* __restrict__ x,
            const _Float16* __restrict__ Wkh,
            const _Float16* __restrict__ Wvh,
            _Float16* __restrict__ Kh,
            _Float16* __restrict__ Vh) {
    // [buf][weight(K/V)][64 rows x 32 halves, padded to WSTR]
    __shared__ __align__(16) _Float16 Wt[2][2][64 * WSTR];

    const int tid  = threadIdx.x;
    const int wave = tid >> 5;
    const int lane = tid & 31;
    const int half = lane >> 4;
    const int ln   = lane & 15;
    const int r0   = blockIdx.x * 128 + wave * 16;
    const int row  = r0 + ln;

    // Staging geometry: 256 threads x (1 K-chunk + 1 V-chunk) of 16B each.
    const int sr = tid >> 2;          // weight row 0..63
    const int sc = (tid & 3) * 8;     // half-offset within 32-wide chunk

    v8f accK[4], accV[4];
#pragma unroll
    for (int j = 0; j < 4; j++) { accK[j] = (v8f){}; accV[j] = (v8f){}; }

    const float* xr = x + (size_t)row * DD;
    const int    c0 = half * 8;       // per-lane A-fragment k offset

    // Prologue: async-stage weight chunk 0, load x chunk 0 into registers.
    async_load_b128((uint32_t)(size_t)&Wt[0][0][sr * WSTR + sc],
                    (uint64_t)(size_t)(Wkh + (size_t)sr * DD + sc));
    async_load_b128((uint32_t)(size_t)&Wt[0][1][sr * WSTR + sc],
                    (uint64_t)(size_t)(Wvh + (size_t)sr * DD + sc));
    v4f f0 = *(const v4f*)(xr + c0);
    v4f f1 = *(const v4f*)(xr + c0 + 4);
    v4f f2 = *(const v4f*)(xr + c0 + 16);
    v4f f3 = *(const v4f*)(xr + c0 + 20);

    int p = 0;
    for (int kk = 0; kk < DD; kk += 32) {
        wait_async0();               // this wave's async writes to buf p done
        __syncthreads();             // all writes visible; prior readers done

        // Kick off next weight chunk into buf p^1 (overlaps the WMMA burst).
        if (kk + 32 < DD) {
            async_load_b128((uint32_t)(size_t)&Wt[p ^ 1][0][sr * WSTR + sc],
                            (uint64_t)(size_t)(Wkh + (size_t)sr * DD + kk + 32 + sc));
            async_load_b128((uint32_t)(size_t)&Wt[p ^ 1][1][sr * WSTR + sc],
                            (uint64_t)(size_t)(Wvh + (size_t)sr * DD + kk + 32 + sc));
        }

        // Build A fragment from the pipelined x registers.
        v16h a;
#pragma unroll
        for (int i = 0; i < 4; i++) {
            a[i]      = (_Float16)f0[i];
            a[4 + i]  = (_Float16)f1[i];
            a[8 + i]  = (_Float16)f2[i];
            a[12 + i] = (_Float16)f3[i];
        }
        // Prefetch next x chunk into the registers just freed.
        if (kk + 32 < DD) {
            const float* xn = xr + kk + 32 + c0;
            f0 = *(const v4f*)(xn);
            f1 = *(const v4f*)(xn + 4);
            f2 = *(const v4f*)(xn + 16);
            f3 = *(const v4f*)(xn + 20);
        }

        // WMMA burst; B fragments from LDS (conflict-free, clause-able).
#pragma unroll
        for (int jn = 0; jn < 4; jn++) {
            const _Float16* kr = &Wt[p][0][(jn * 16 + ln) * WSTR + c0];
            accK[jn] = __builtin_amdgcn_wmma_f32_16x16x32_f16(
                false, a, false, make_frag(*(const v8h*)kr, *(const v8h*)(kr + 16)),
                (short)0, accK[jn], false, false);
            const _Float16* vr = &Wt[p][1][(jn * 16 + ln) * WSTR + c0];
            accV[jn] = __builtin_amdgcn_wmma_f32_16x16x32_f16(
                false, a, false, make_frag(*(const v8h*)vr, *(const v8h*)(vr + 16)),
                (short)0, accV[jn], false, false);
        }
        p ^= 1;
    }

    // Store C-layout accumulators: VGPR j -> row (j + 8*half), lane -> col ln.
#pragma unroll
    for (int jn = 0; jn < 4; jn++) {
#pragma unroll
        for (int j = 0; j < 8; j++) {
            const int m   = r0 + j + 8 * half;
            const int col = jn * 16 + ln;
            Kh[(size_t)m * HSZ + col] = (_Float16)accK[jn][j];
            Vh[(size_t)m * HSZ + col] = (_Float16)accV[jn][j];
        }
    }
}

// ---------------------------------------------------------------------------
// Kernel 2: causal flash attention.  q == v == Vh per reference.
// 128 threads = 4 waves/block; block owns 64 query rows (16 per wave).
// K tiles: double-buffered GLOBAL_LOAD_ASYNC_TO_LDS_B128 (ASYNCcnt).
// V tiles: software-pipelined through VGPRs, stored transposed into LDS.
// Softmax runs base-2; the denominator is accumulated by WMMA against a
// constant ones-column B fragment (no row-sum shuffle butterflies).
// ---------------------------------------------------------------------------
__global__ void __launch_bounds__(128)
attn_kernel(const _Float16* __restrict__ Kh,
            const _Float16* __restrict__ Vh,
            float* __restrict__ out) {
    __shared__ __align__(16) _Float16 Kt[2][64 * KSTR];   // [key][h], async-filled
    __shared__ __align__(16) _Float16 Vt[2][64 * KSTR];   // [h][key]  (transposed)
    __shared__ __align__(16) _Float16 Ps[4][16 * KSTR];   // per-wave P staging

    const int tid  = threadIdx.x;
    const int wave = tid >> 5;
    const int lane = tid & 31;
    const int half = lane >> 4;
    const int ln   = lane & 15;

    // Heavy (diagonal) q-tiles dispatch first for better backfill.
    const int nqt   = TT / 64;
    const int b     = blockIdx.x % BB;
    const int qt    = nqt - 1 - (blockIdx.x / BB);
    const int qBase = qt * 64;
    const int qr0   = qBase + wave * 16;

    const _Float16* Kb = Kh + (size_t)b * TT * HSZ;
    const _Float16* Vb = Vh + (size_t)b * TT * HSZ;

    // Per-thread staging chunk geometry: 512 8-half chunks, 4 per thread.
    int ckey[4], ch0[4];
#pragma unroll
    for (int i = 0; i < 4; i++) {
        const int ch = tid + i * 128;
        ckey[i] = ch >> 3;
        ch0[i]  = (ch & 7) * 8;
    }

    // Q fragments (two 16x32 K-chunks covering HS=64).
    v16h qa[2];
    {
        const _Float16* qrow = Vb + (size_t)(qr0 + ln) * HSZ;
#pragma unroll
        for (int kc = 0; kc < 2; kc++) {
            const int c0 = kc * 32 + half * 8;
            qa[kc] = make_frag(*(const v8h*)(qrow + c0), *(const v8h*)(qrow + c0 + 16));
        }
    }

    // Constant ones-column B fragment: column n=0 is 1.0 for every k.
    v16h bones;
#pragma unroll
    for (int i = 0; i < 16; i++) bones[i] = (ln == 0) ? (_Float16)1.0f : (_Float16)0.0f;

    v8f O[4];
#pragma unroll
    for (int j = 0; j < 4; j++) O[j] = (v8f){};
    v8f Osum = (v8f){};              // column 0 = running softmax denominator
    float rowM[8];
#pragma unroll
    for (int j = 0; j < 8; j++) rowM[j] = -1.0e30f;

    // scale * log2(e): softmax runs base-2.
    const float kScale = 0.125f * 1.44269504088896340736f;

    const int nkt = qt + 1;          // causal: key tiles up to the diagonal
    int p = 0;
    v8h vreg[4];

    // Prologue: start staging tile 0.
#pragma unroll
    for (int i = 0; i < 4; i++) {
        async_load_b128((uint32_t)(size_t)&Kt[0][ckey[i] * KSTR + ch0[i]],
                        (uint64_t)(size_t)(Kb + (size_t)ckey[i] * HSZ + ch0[i]));
        vreg[i] = *(const v8h*)(Vb + (size_t)ckey[i] * HSZ + ch0[i]);
    }

    for (int kt = 0; kt < nkt; kt++) {
        const int kBase = kt * 64;

        // Finish staging buffer p: K via ASYNCcnt, V via transpose-store.
        wait_async0();
#pragma unroll
        for (int i = 0; i < 4; i++) {
#pragma unroll
            for (int e = 0; e < 8; e++)
                Vt[p][(ch0[i] + e) * KSTR + ckey[i]] = vreg[i][e];
        }
        __syncthreads();

        // Kick off tile kt+1 into buffer p^1; overlaps with compute below.
        if (kt + 1 < nkt) {
            const int nBase = kBase + 64;
#pragma unroll
            for (int i = 0; i < 4; i++) {
                async_load_b128((uint32_t)(size_t)&Kt[p ^ 1][ckey[i] * KSTR + ch0[i]],
                                (uint64_t)(size_t)(Kb + (size_t)(nBase + ckey[i]) * HSZ + ch0[i]));
                vreg[i] = *(const v8h*)(Vb + (size_t)(nBase + ckey[i]) * HSZ + ch0[i]);
            }
        }

        // ---- S = Q K^T (base-2 scaled), causal-masked; 4 n-tiles of 16 keys.
        v8f  s4[4];
        float tmax[8];
#pragma unroll
        for (int j = 0; j < 8; j++) tmax[j] = -1.0e30f;

#pragma unroll
        for (int jn = 0; jn < 4; jn++) {
            v8f s = (v8f){};
#pragma unroll
            for (int kc = 0; kc < 2; kc++) {
                const _Float16* kr = &Kt[p][(jn * 16 + ln) * KSTR + kc * 32 + half * 8];
                v16h bk = make_frag(*(const v8h*)kr, *(const v8h*)(kr + 16));
                s = __builtin_amdgcn_wmma_f32_16x16x32_f16(
                    false, qa[kc], false, bk, (short)0, s, false, false);
            }
            const int kidx = kBase + jn * 16 + ln;
#pragma unroll
            for (int j = 0; j < 8; j++) {
                const int qidx = qr0 + j + 8 * half;
                float v = s[j] * kScale;
                v = (kidx <= qidx) ? v : -1.0e30f;
                s[j] = v;
                tmax[j] = fmaxf(tmax[j], v);
            }
            s4[jn] = s;
        }

        // ---- Online softmax (base-2): row max across the 16 lanes of a half.
        float alpha[8];
#pragma unroll
        for (int j = 0; j < 8; j++) {
            float m = tmax[j];
#pragma unroll
            for (int off = 1; off < 16; off <<= 1)
                m = fmaxf(m, __shfl_xor(m, off, 32));
            const float mNew = fmaxf(rowM[j], m);
            alpha[j] = exp2f(rowM[j] - mNew);
            rowM[j]  = mNew;
        }

        // P = 2^(S - M); rescale O and the denominator tile by alpha.
#pragma unroll
        for (int j = 0; j < 8; j++) {
#pragma unroll
            for (int jn = 0; jn < 4; jn++)
                s4[jn][j] = exp2f(s4[jn][j] - rowM[j]);
            Osum[j] *= alpha[j];
#pragma unroll
            for (int jn = 0; jn < 4; jn++) O[jn][j] *= alpha[j];
        }

        // ---- Stage P (f16) through wave-private LDS: C-layout -> A-fragment.
        _Float16* Pw = &Ps[wave][0];
#pragma unroll
        for (int jn = 0; jn < 4; jn++)
#pragma unroll
            for (int j = 0; j < 8; j++)
                Pw[(j + 8 * half) * KSTR + jn * 16 + ln] = (_Float16)s4[jn][j];

        v16h pa[2];
        {
            const _Float16* prow = Pw + ln * KSTR;
#pragma unroll
            for (int kc = 0; kc < 2; kc++) {
                const int c0 = kc * 32 + half * 8;
                pa[kc] = make_frag(*(const v8h*)(prow + c0), *(const v8h*)(prow + c0 + 16));
            }
        }

        // ---- O += P * V ; denominator += P * ones-column.
#pragma unroll
        for (int jn = 0; jn < 4; jn++) {
#pragma unroll
            for (int kc = 0; kc < 2; kc++) {
                const _Float16* vr = &Vt[p][(jn * 16 + ln) * KSTR + kc * 32 + half * 8];
                v16h bv = make_frag(*(const v8h*)vr, *(const v8h*)(vr + 16));
                O[jn] = __builtin_amdgcn_wmma_f32_16x16x32_f16(
                    false, pa[kc], false, bv, (short)0, O[jn], false, false);
            }
        }
#pragma unroll
        for (int kc = 0; kc < 2; kc++)
            Osum = __builtin_amdgcn_wmma_f32_16x16x32_f16(
                false, pa[kc], false, bones, (short)0, Osum, false, false);

        p ^= 1;
    }

    // Epilogue: L_j lives in column 0 (lane 0 / lane 16); broadcast, divide, store.
#pragma unroll
    for (int j = 0; j < 8; j++) {
        const float Lj  = __shfl(Osum[j], half * 16, 32);
        const float inv = 1.0f / Lj;
        const int   m   = qr0 + j + 8 * half;
#pragma unroll
        for (int jn = 0; jn < 4; jn++) {
            out[((size_t)b * TT + m) * HSZ + jn * 16 + ln] = O[jn][j] * inv;
        }
    }
}

// ---------------------------------------------------------------------------
// Host launch.
// ---------------------------------------------------------------------------
extern "C" void kernel_launch(void* const* d_in, const int* in_sizes, int n_in,
                              void* d_out, int out_size, void* d_ws, size_t ws_size,
                              hipStream_t stream) {
    const float* x  = (const float*)d_in[0];
    const float* Wk = (const float*)d_in[1];
    const float* Wv = (const float*)d_in[2];
    float* out = (float*)d_out;

    char* w = (char*)d_ws;
    _Float16* Kh  = (_Float16*)(w);                                 // 2 MB
    _Float16* Vh  = (_Float16*)(w + (size_t)2 * 1024 * 1024);       // 2 MB
    _Float16* Wkh = (_Float16*)(w + (size_t)4 * 1024 * 1024);       // 128 KB
    _Float16* Wvh = (_Float16*)(w + (size_t)4 * 1024 * 1024 + 128 * 1024);

    (void)in_sizes; (void)n_in; (void)out_size; (void)ws_size;

    // 0) weights -> f16
    cvt_w_kernel<<<(HSZ * DD + 255) / 256, 256, 0, stream>>>(Wk, Wv, Wkh, Wvh);
    // 1) projections (WMMA, async LDS-staged weights)
    proj_kernel<<<(BB * TT) / 128, 256, 0, stream>>>(x, Wkh, Wvh, Kh, Vh);
    // 2) causal flash attention (WMMA + async-to-LDS double buffering)
    attn_kernel<<<BB * (TT / 64), 128, 0, stream>>>(Kh, Vh, out);
}
